// MixtralTransformerDecoder_34067680592360
// MI455X (gfx1250) — compile-verified, matrix operand order checked by reading, not run on hardware
//
#include <hip/hip_runtime.h>
#include <hip/hip_bf16.h>
#include <math.h>

typedef __bf16 bf16;
typedef __attribute__((ext_vector_type(16))) __bf16 v16bf;
typedef __attribute__((ext_vector_type(8)))  float  v8f;

#define T_TOK 4096
#define HDIM  1024
#define IDIM  2048
#define NEXP  8
#define BM    128
#define BN    64
#define BK    32

// ---------------------------------------------------------------------------
// helpers
// ---------------------------------------------------------------------------
static __device__ __forceinline__ unsigned pack_bf16(float a, float b) {
  unsigned short ua = __builtin_bit_cast(unsigned short, (bf16)a);
  unsigned short ub = __builtin_bit_cast(unsigned short, (bf16)b);
  return (unsigned)ua | ((unsigned)ub << 16);
}

// First K index of the bf16 pair held in "VGPR" j of a 16x32 A fragment
// (ISA 7.12.2, 16-bit A-matrix layout); B 32x16 uses the mirrored mapping.
static __device__ __forceinline__ int kmap(int half, int j) {
  return (j < 4) ? (half * 8 + j * 2) : (16 + half * 8 + (j - 4) * 2);
}

static __device__ __forceinline__ v8f wmma_bf16(v16bf a, v16bf b, v8f c) {
  return __builtin_amdgcn_wmma_f32_16x16x32_bf16(
      /*neg_a=*/false, a, /*neg_b=*/false, b,
      /*c_mod=*/(short)0, c, /*reuse_a=*/false, /*reuse_b=*/false);
}

// Async global->LDS copy of one DWORD, GVS form (ASYNCcnt):
//   LDS[vdst_lds_addr] = MEM[s_base + v_offset]
// Per-lane LDS dest and per-lane 32-bit offset are loop-invariant; only the
// scalar base advances each K-step.
static __device__ __forceinline__ void async_copy_b32_s(unsigned lds_addr,
                                                        unsigned voff,
                                                        const void* sbase) {
  asm volatile("global_load_async_to_lds_b32 %0, %1, %2"
               :
               : "v"(lds_addr), "v"(voff),
                 "s"((unsigned long long)(uintptr_t)sbase)
               : "memory");
}

static __device__ __forceinline__ void wait_asynccnt0() {
  asm volatile("s_wait_asynccnt 0x0" ::: "memory");
}

// ---------------------------------------------------------------------------
// Router: logits = X @ RK, softmax, top-2, renormalize -> routing[E][T].
// Raw logits also go to the second output of the tuple.
// ---------------------------------------------------------------------------
__global__ __launch_bounds__(256)
void router_kernel(const float* __restrict__ X,       // [T, H]
                   const float* __restrict__ RK,      // [H, E]
                   float* __restrict__ routing,       // [E, T]
                   float* __restrict__ logits_out) {  // [T, E]
  const int wave = threadIdx.x >> 5;
  const int lane = threadIdx.x & 31;
  const int t = blockIdx.x * 8 + wave;
  if (t >= T_TOK) return;

  float acc[NEXP];
#pragma unroll
  for (int e = 0; e < NEXP; ++e) acc[e] = 0.f;

  for (int h = lane; h < HDIM; h += 32) {
    float x = X[(size_t)t * HDIM + h];
#pragma unroll
    for (int e = 0; e < NEXP; ++e) acc[e] += x * RK[h * NEXP + e];
  }
#pragma unroll
  for (int e = 0; e < NEXP; ++e) {
#pragma unroll
    for (int s = 16; s >= 1; s >>= 1) acc[e] += __shfl_xor(acc[e], s, 32);
  }

  if (lane == 0) {
    float m = acc[0];
#pragma unroll
    for (int e = 1; e < NEXP; ++e) m = fmaxf(m, acc[e]);
    float p[NEXP], sum = 0.f;
#pragma unroll
    for (int e = 0; e < NEXP; ++e) { p[e] = __expf(acc[e] - m); sum += p[e]; }
#pragma unroll
    for (int e = 0; e < NEXP; ++e) p[e] /= sum;

    int i1 = 0;
#pragma unroll
    for (int e = 1; e < NEXP; ++e) if (p[e] > p[i1]) i1 = e;
    int i2 = (i1 == 0) ? 1 : 0;
#pragma unroll
    for (int e = 0; e < NEXP; ++e) if (e != i1 && p[e] > p[i2]) i2 = e;
    float s2 = p[i1] + p[i2];

#pragma unroll
    for (int e = 0; e < NEXP; ++e) {
      float w = (e == i1) ? p[i1] / s2 : (e == i2) ? p[i2] / s2 : 0.f;
      routing[(size_t)e * T_TOK + t] = w;
      logits_out[(size_t)t * NEXP + e] = acc[e];
    }
  }
}

// ---------------------------------------------------------------------------
// fp32 -> bf16 conversion of the activations (done once).
// ---------------------------------------------------------------------------
__global__ __launch_bounds__(256)
void cvt_bf16_kernel(const float* __restrict__ src, bf16* __restrict__ dst, int n) {
  int i = (blockIdx.x * blockDim.x + threadIdx.x) * 4;
  if (i + 3 < n) {
    float4 v = *(const float4*)(src + i);
    unsigned* d = (unsigned*)(dst + i);
    d[0] = pack_bf16(v.x, v.y);
    d[1] = pack_bf16(v.z, v.w);
  }
}

// ---------------------------------------------------------------------------
// B-tile staging helper: thread handles one aligned row pair (k0, k0+1) and
// 4 columns of a 32xBN fp32 weight tile; loads two coalesced float4 rows,
// packs to bf16 pairs, scatters 4 DWORDs into the fragment-major LDS image.
//   half = (k0>>3)&1, j = ((k0&7)>>1) + ((k0>>4)&1)*4  (inverse of kmap)
// ---------------------------------------------------------------------------
static __device__ __forceinline__ void stage_B_rowpair(unsigned* lds,
                                                       const float* __restrict__ W,
                                                       size_t row_off, size_t stride,
                                                       int k0, int cb) {
  const float4 r0 = *(const float4*)(W + row_off);
  const float4 r1 = *(const float4*)(W + row_off + stride);
  const int half = (k0 >> 3) & 1;
  const int j = ((k0 & 7) >> 1) + ((k0 >> 4) & 1) * 4;
  unsigned u[4];
  u[0] = pack_bf16(r0.x, r1.x);
  u[1] = pack_bf16(r0.y, r1.y);
  u[2] = pack_bf16(r0.z, r1.z);
  u[3] = pack_bf16(r0.w, r1.w);
#pragma unroll
  for (int c = 0; c < 4; ++c) {
    int nloc = cb + c;
    int sb = nloc >> 4;
    int L = (nloc & 15) | (half << 4);
    lds[(sb * 32 + L) * 8 + j] = u[c];
  }
}

// ---------------------------------------------------------------------------
// Expert gate/inter GEMM: hidden[t,i] = silu(X@Wg) * (X@Wi), bf16 output.
// 256 threads = 8 waves; 128x64 tile; each wave owns a 32x32 subtile of BOTH
// products (sharing A fragments). A tile staged by async global->LDS DWORD
// copies (GVS form, loop-invariant lane offsets); B tiles staged by coalesced
// float4 row loads + v_cvt_pk_bf16 + LDS scatter.
// ---------------------------------------------------------------------------
__global__ __launch_bounds__(256)
void moe_gateinter_kernel(const bf16*  __restrict__ Xb,  // [T, H] bf16
                          const float* __restrict__ Wg,  // [E, H, I]
                          const float* __restrict__ Wi,  // [E, H, I]
                          bf16*        __restrict__ Hid, // [T, I] bf16
                          int e) {
  __shared__ __align__(16) unsigned ldsA [8 * 32 * 8];  // 128x32 bf16, frag-major
  __shared__ __align__(16) unsigned ldsBg[4 * 32 * 8];  //  32x64 bf16, frag-major
  __shared__ __align__(16) unsigned ldsBi[4 * 32 * 8];

  const int tid  = threadIdx.x;
  const int lane = tid & 31;
  const int wv   = tid >> 5;
  const int m0   = blockIdx.x * BM;
  const int n0   = blockIdx.y * BN;
  const int wm   = (wv & 3) * 32;   // wave row offset in block tile
  const int wn   = (wv >> 2) * 32;  // wave col offset in block tile

  const size_t wbase = (size_t)e * HDIM * IDIM;

  v8f accG[2][2], accI[2][2];
#pragma unroll
  for (int a = 0; a < 2; ++a)
#pragma unroll
    for (int b = 0; b < 2; ++b) { accG[a][b] = 0.f; accI[a][b] = 0.f; }

  // ---- A staging coordinates: 8 sub-blocks x 32 lane-slots ----
  const int sA = tid >> 5, LA = tid & 31, hA = LA >> 4;
  const int rowA = m0 + sA * 16 + (LA & 15);
  const unsigned ldsA_base = (unsigned)(uintptr_t)&ldsA[(sA * 32 + LA) * 8];
  unsigned aoff[8];  // loop-invariant per-lane byte offsets into Xb[., kk..]
#pragma unroll
  for (int j = 0; j < 8; ++j)
    aoff[j] = (unsigned)(((size_t)rowA * HDIM + kmap(hA, j)) * sizeof(bf16));

  // ---- B staging coordinates: row pair + 4 columns per thread ----
  const int k0 = (tid >> 4) * 2;      // 0..30
  const int cb = (tid & 15) * 4;      // 0..60

  for (int kk = 0; kk < HDIM; kk += BK) {
    // A tile: 8 async DWORD copies, scalar base advances with kk
#pragma unroll
    for (int j = 0; j < 8; ++j)
      async_copy_b32_s(ldsA_base + j * 4, aoff[j], Xb + kk);

    // B tiles: coalesced row loads, convert, scatter to frag-major LDS
    size_t roff = wbase + (size_t)(kk + k0) * IDIM + n0 + cb;
    stage_B_rowpair(ldsBg, Wg, roff, IDIM, k0, cb);
    stage_B_rowpair(ldsBi, Wi, roff, IDIM, k0, cb);

    // prefetch next weight K-slab toward L2
    if (kk + BK < HDIM) {
      __builtin_prefetch(Wg + roff + (size_t)BK * IDIM, 0, 1);
      __builtin_prefetch(Wi + roff + (size_t)BK * IDIM, 0, 1);
    }
    wait_asynccnt0();
    __syncthreads();

    // ---- WMMA: 2x2 tiles for gate and inter, shared A fragments ----
    const v16bf* fA = (const v16bf*)ldsA;
    const v16bf* fG = (const v16bf*)ldsBg;
    const v16bf* fI = (const v16bf*)ldsBi;
    const int sa = wm >> 4, sb0 = wn >> 4;
    v16bf a0  = fA[(sa + 0) * 32 + lane];
    v16bf a1  = fA[(sa + 1) * 32 + lane];
    v16bf bg0 = fG[(sb0 + 0) * 32 + lane];
    v16bf bg1 = fG[(sb0 + 1) * 32 + lane];
    v16bf bi0 = fI[(sb0 + 0) * 32 + lane];
    v16bf bi1 = fI[(sb0 + 1) * 32 + lane];

    accG[0][0] = wmma_bf16(a0, bg0, accG[0][0]);
    accG[0][1] = wmma_bf16(a0, bg1, accG[0][1]);
    accG[1][0] = wmma_bf16(a1, bg0, accG[1][0]);
    accG[1][1] = wmma_bf16(a1, bg1, accG[1][1]);
    accI[0][0] = wmma_bf16(a0, bi0, accI[0][0]);
    accI[0][1] = wmma_bf16(a0, bi1, accI[0][1]);
    accI[1][0] = wmma_bf16(a1, bi0, accI[1][0]);
    accI[1][1] = wmma_bf16(a1, bi1, accI[1][1]);
    __syncthreads();
  }

  // ---- epilogue: hidden = silu(gate) * inter, store bf16 ----
#pragma unroll
  for (int dm = 0; dm < 2; ++dm)
#pragma unroll
    for (int dn = 0; dn < 2; ++dn)
#pragma unroll
      for (int v = 0; v < 8; ++v) {
        float g  = accG[dm][dn][v];
        float it = accI[dm][dn][v];
        float hv = it * (g / (1.f + __expf(-g)));
        int row = m0 + wm + dm * 16 + v + ((lane >> 4) << 3);
        int col = n0 + wn + dn * 16 + (lane & 15);
        Hid[(size_t)row * IDIM + col] = (bf16)hv;
      }
}

// ---------------------------------------------------------------------------
// Expert out GEMM: out[t,h] (+)= routing[e,t] * (hidden @ Wo[e]).
// e == 0 writes, e > 0 accumulates (kernels are stream-ordered per expert).
// ---------------------------------------------------------------------------
__global__ __launch_bounds__(256)
void moe_out_kernel(const bf16*  __restrict__ Hid,     // [T, I] bf16
                    const float* __restrict__ Wo,      // [E, I, H]
                    const float* __restrict__ routing, // [E, T]
                    float*       __restrict__ Out,     // [T, H]
                    int e) {
  __shared__ __align__(16) unsigned ldsA[8 * 32 * 8];  // 128x32 bf16
  __shared__ __align__(16) unsigned ldsB[4 * 32 * 8];  //  32x64 bf16

  const int tid  = threadIdx.x;
  const int lane = tid & 31;
  const int wv   = tid >> 5;
  const int m0   = blockIdx.x * BM;
  const int n0   = blockIdx.y * BN;
  const int wm   = (wv & 3) * 32;
  const int wn   = (wv >> 2) * 32;

  const size_t wbase = (size_t)e * IDIM * HDIM;

  v8f acc[2][2];
#pragma unroll
  for (int a = 0; a < 2; ++a)
#pragma unroll
    for (int b = 0; b < 2; ++b) acc[a][b] = 0.f;

  const int sA = tid >> 5, LA = tid & 31, hA = LA >> 4;
  const int rowA = m0 + sA * 16 + (LA & 15);
  const unsigned ldsA_base = (unsigned)(uintptr_t)&ldsA[(sA * 32 + LA) * 8];
  unsigned aoff[8];
#pragma unroll
  for (int j = 0; j < 8; ++j)
    aoff[j] = (unsigned)(((size_t)rowA * IDIM + kmap(hA, j)) * sizeof(bf16));

  const int k0 = (tid >> 4) * 2;
  const int cb = (tid & 15) * 4;

  for (int kk = 0; kk < IDIM; kk += BK) {
#pragma unroll
    for (int j = 0; j < 8; ++j)
      async_copy_b32_s(ldsA_base + j * 4, aoff[j], Hid + kk);

    size_t roff = wbase + (size_t)(kk + k0) * HDIM + n0 + cb;
    stage_B_rowpair(ldsB, Wo, roff, HDIM, k0, cb);

    if (kk + BK < IDIM)
      __builtin_prefetch(Wo + roff + (size_t)BK * HDIM, 0, 1);
    wait_asynccnt0();
    __syncthreads();

    const v16bf* fA = (const v16bf*)ldsA;
    const v16bf* fB = (const v16bf*)ldsB;
    const int sa = wm >> 4, sb0 = wn >> 4;
    v16bf a0 = fA[(sa + 0) * 32 + lane];
    v16bf a1 = fA[(sa + 1) * 32 + lane];
    v16bf b0 = fB[(sb0 + 0) * 32 + lane];
    v16bf b1 = fB[(sb0 + 1) * 32 + lane];

    acc[0][0] = wmma_bf16(a0, b0, acc[0][0]);
    acc[0][1] = wmma_bf16(a0, b1, acc[0][1]);
    acc[1][0] = wmma_bf16(a1, b0, acc[1][0]);
    acc[1][1] = wmma_bf16(a1, b1, acc[1][1]);
    __syncthreads();
  }

#pragma unroll
  for (int dm = 0; dm < 2; ++dm)
#pragma unroll
    for (int dn = 0; dn < 2; ++dn)
#pragma unroll
      for (int v = 0; v < 8; ++v) {
        int row = m0 + wm + dm * 16 + v + ((lane >> 4) << 3);
        int col = n0 + wn + dn * 16 + (lane & 15);
        float w = routing[(size_t)e * T_TOK + row];
        float val = acc[dm][dn][v] * w;
        size_t o = (size_t)row * HDIM + col;
        if (e == 0) Out[o] = val;
        else        Out[o] += val;
      }
}

// ---------------------------------------------------------------------------
// launch
// ---------------------------------------------------------------------------
extern "C" void kernel_launch(void* const* d_in, const int* in_sizes, int n_in,
                              void* d_out, int out_size, void* d_ws, size_t ws_size,
                              hipStream_t stream) {
  const float* X  = (const float*)d_in[0];  // [2,2048,1024] -> [T,H]
  const float* RK = (const float*)d_in[1];  // [H,E]
  const float* Wg = (const float*)d_in[2];  // [E,H,I]
  const float* Wi = (const float*)d_in[3];  // [E,H,I]
  const float* Wo = (const float*)d_in[4];  // [E,I,H]

  float* out    = (float*)d_out;                       // [T*H] fp32
  float* logits = out + (size_t)T_TOK * HDIM;          // [T*E] fp32 (tuple part 2)

  char* ws = (char*)d_ws;
  float* routing = (float*)ws;                                       // E*T f32   (128 KB)
  bf16*  Xb      = (bf16*)(ws + (size_t)128 * 1024);                 // T*H bf16  (8 MB)
  bf16*  Hid     = (bf16*)(ws + (size_t)128 * 1024 +
                           (size_t)T_TOK * HDIM * sizeof(bf16));     // T*I bf16  (16 MB)

  router_kernel<<<T_TOK / 8, 256, 0, stream>>>(X, RK, routing, logits);
  cvt_bf16_kernel<<<(T_TOK * HDIM / 4 + 255) / 256, 256, 0, stream>>>(X, Xb, T_TOK * HDIM);

  dim3 gridGI(T_TOK / BM, IDIM / BN);  // 32 x 32
  dim3 gridO (T_TOK / BM, HDIM / BN);  // 32 x 16
  for (int e = 0; e < NEXP; ++e) {
    moe_gateinter_kernel<<<gridGI, 256, 0, stream>>>(Xb, Wg, Wi, Hid, e);
    moe_out_kernel<<<gridO, 256, 0, stream>>>(Hid, Wo, routing, out, e);
  }
}
